// SparseGaussianProcess_35201551958592
// MI455X (gfx1250) — matrix operand order repaired and checked by reading, not compiled, not run on hardware
//
#include <hip/hip_runtime.h>
#include <hip/hip_bf16.h>

typedef __attribute__((ext_vector_type(16))) _Float16 v16h;
typedef __attribute__((ext_vector_type(8)))  _Float16 v8h;
typedef __attribute__((ext_vector_type(8)))  float    v8f;
typedef __attribute__((ext_vector_type(2)))  float    v2f;

#define N_PTS  16384
#define IDIM   8
#define M_IND  1024
#define L_FEAT 1024
#define S_SMP  8
#define ODIM   4

// out[s][o][n] = f_prior + f_data, column c = s*ODIM + o in [0,32)
__global__ __launch_bounds__(256) void sgp_fused_kernel(
    const float* __restrict__ x,     // [N, 8]
    const float* __restrict__ zloc,  // [1024, 8]
    const float* __restrict__ iw,    // [8, 4, 1024] = [c, m]
    const float* __restrict__ pf,    // [4, 8, 1024] = [o, i, l]
    const float* __restrict__ pph,   // [4, 1024]
    const float* __restrict__ pw,    // [8, 4, 1024] = [c, l]
    float* __restrict__ out)         // [32, N]
{
    __shared__ float    s_vec[M_IND * IDIM];  // zloc rows, then freq_o transposed [l][i]
    __shared__ float    s_sc[M_IND];          // |z|^2, then phase_o
    __shared__ float    s_stage[8][16][32];   // per-wave output tile [row][col]
    __shared__ float    s_xsq[8][16];         // per-wave |x_row|^2
    __shared__ _Float16 s_kst[8][16][32];     // per-wave f16 re-layout stage [row][k]

    const int tid   = threadIdx.x;
    const int lane  = tid & 31;
    const int wv    = tid >> 5;
    const int col   = lane & 15;   // D column / A row owned by this lane
    const int khalf = lane >> 4;
    const int nbase = (blockIdx.x * 8 + wv) * 16;

    // ---- cooperative: inducing locations + squared norms into LDS ----
    for (int m = tid; m < M_IND; m += 256) {
        float ssum = 0.f;
        #pragma unroll
        for (int i = 0; i < IDIM; ++i) {
            float v = zloc[m * IDIM + i];
            s_vec[m * IDIM + i] = v;
            ssum += v * v;
        }
        s_sc[m] = ssum;
    }
    __syncthreads();

    // ---- per-lane x row (row = lane % 16 of this wave's tile) ----
    const int n = nbase + col;
    float xv[IDIM];
    #pragma unroll
    for (int i = 0; i < IDIM; ++i) xv[i] = x[n * IDIM + i];
    float xsq = 0.f;
    #pragma unroll
    for (int i = 0; i < IDIM; ++i) xsq = fmaf(xv[i], xv[i], xsq);
    s_xsq[wv][col] = xsq;  // DS in-order within wave; read back below

    // f32 WMMA A fragments for X[16x8]: lane half selects K pair within each K=4 step
    const v2f xa0 = { xv[2 * khalf],     xv[2 * khalf + 1]     };  // K = 0..3
    const v2f xa1 = { xv[4 + 2 * khalf], xv[4 + 2 * khalf + 1] };  // K = 4..7

    // squared norms of the 8 rows this lane owns in D layout (row = khalf*8 + r)
    float xsqr[8];
    #pragma unroll
    for (int r = 0; r < 8; ++r) xsqr[r] = s_xsq[wv][khalf * 8 + r];

    // ================= data phase: K(x,z) @ inducing_weights =================
    v8f acc0 = {};   // columns 0..15  (c = s*4+o)
    v8f acc1 = {};   // columns 16..31
    for (int kc = 0; kc < M_IND / 32; ++kc) {
        const int k0 = kc * 32;

        // two 16x16 inner-product tiles via f32 WMMA, exp in D layout, stage as f16 [row][m]
        #pragma unroll
        for (int t = 0; t < 2; ++t) {
            const int mcol = k0 + t * 16 + col;
            const float* zp = &s_vec[mcol * IDIM];
            const v2f zb0 = { zp[2 * khalf],     zp[2 * khalf + 1]     };
            const v2f zb1 = { zp[4 + 2 * khalf], zp[4 + 2 * khalf + 1] };
            v8f dacc = {};
            dacc = __builtin_amdgcn_wmma_f32_16x16x4_f32(false, xa0, false, zb0,
                                                         (short)0, dacc, false, false);
            dacc = __builtin_amdgcn_wmma_f32_16x16x4_f32(false, xa1, false, zb1,
                                                         (short)0, dacc, false, false);
            const float zsq = s_sc[mcol];
            #pragma unroll
            for (int r = 0; r < 8; ++r) {
                const float dist = xsqr[r] + zsq - 2.f * dacc[r];
                s_kst[wv][khalf * 8 + r][t * 16 + col] = (_Float16)__expf(-0.5f * dist);
            }
        }

        // reload as f16 WMMA A fragment: pairs packed along K, contiguous in stage rows
        const v8h alo = *(const v8h*)&s_kst[wv][col][khalf * 16];      // K = khalf*16 + 0..7
        const v8h ahi = *(const v8h*)&s_kst[wv][col][khalf * 16 + 8];  // K = khalf*16 + 8..15
        const v16h a = __builtin_shufflevector(alo, ahi, 0, 1, 2, 3, 4, 5, 6, 7,
                                               8, 9, 10, 11, 12, 13, 14, 15);

        v16h b0, b1;
        {
            const float* wp0 = iw + (0 * 16 + col) * M_IND + k0 + khalf * 16;
            const float* wp1 = iw + (1 * 16 + col) * M_IND + k0 + khalf * 16;
            #pragma unroll
            for (int j = 0; j < 16; ++j) {
                b0[j] = (_Float16)wp0[j];
                b1[j] = (_Float16)wp1[j];
            }
        }

        acc0 = __builtin_amdgcn_wmma_f32_16x16x32_f16(false, a, false, b0,
                                                      (short)0, acc0, false, false);
        acc1 = __builtin_amdgcn_wmma_f32_16x16x32_f16(false, a, false, b1,
                                                      (short)0, acc1, false, false);
    }

    // stage f_data tile into this wave's LDS region (D layout: row = khalf*8+r)
    #pragma unroll
    for (int r = 0; r < 8; ++r) {
        const int rr = khalf * 8 + r;
        s_stage[wv][rr][col]      = acc0[r];
        s_stage[wv][rr][16 + col] = acc1[r];
    }

    // ================= prior phase: cos(xF+phi) @ sqrt(2/L)*w =================
    const float bscale = 0.04419417382415922f;  // sqrt(2/1024)
    for (int o = 0; o < ODIM; ++o) {
        __syncthreads();  // everyone done with previous s_vec contents
        for (int l = tid; l < L_FEAT; l += 256) {
            #pragma unroll
            for (int i = 0; i < IDIM; ++i)
                s_vec[l * IDIM + i] = pf[(o * IDIM + i) * L_FEAT + l];
            s_sc[l] = pph[o * L_FEAT + l];
        }
        __syncthreads();

        v8f accp = {};
        for (int kc = 0; kc < L_FEAT / 32; ++kc) {
            const int k0 = kc * 32;

            #pragma unroll
            for (int t = 0; t < 2; ++t) {
                const int lcol = k0 + t * 16 + col;
                const float* fp = &s_vec[lcol * IDIM];
                const v2f fb0 = { fp[2 * khalf],     fp[2 * khalf + 1]     };
                const v2f fb1 = { fp[4 + 2 * khalf], fp[4 + 2 * khalf + 1] };
                const float ph = s_sc[lcol];
                v8f dacc;  // fold phase bias into the WMMA accumulator
                #pragma unroll
                for (int r = 0; r < 8; ++r) dacc[r] = ph;
                dacc = __builtin_amdgcn_wmma_f32_16x16x4_f32(false, xa0, false, fb0,
                                                             (short)0, dacc, false, false);
                dacc = __builtin_amdgcn_wmma_f32_16x16x4_f32(false, xa1, false, fb1,
                                                             (short)0, dacc, false, false);
                #pragma unroll
                for (int r = 0; r < 8; ++r) {
                    s_kst[wv][khalf * 8 + r][t * 16 + col] = (_Float16)__cosf(dacc[r]);
                }
            }

            const v8h alo = *(const v8h*)&s_kst[wv][col][khalf * 16];
            const v8h ahi = *(const v8h*)&s_kst[wv][col][khalf * 16 + 8];
            const v16h a = __builtin_shufflevector(alo, ahi, 0, 1, 2, 3, 4, 5, 6, 7,
                                                   8, 9, 10, 11, 12, 13, 14, 15);

            v16h b;
            if (col < S_SMP) {
                const float* wp = pw + (col * ODIM + o) * L_FEAT + k0 + khalf * 16;
                #pragma unroll
                for (int j = 0; j < 16; ++j) b[j] = (_Float16)(bscale * wp[j]);
            } else {
                #pragma unroll
                for (int j = 0; j < 16; ++j) b[j] = (_Float16)0.f;
            }

            accp = __builtin_amdgcn_wmma_f32_16x16x32_f16(false, a, false, b,
                                                          (short)0, accp, false, false);
        }

        // fold f_prior for this o into the staged tile (col s -> c = s*4+o)
        if (col < S_SMP) {
            const int c = col * ODIM + o;
            #pragma unroll
            for (int r = 0; r < 8; ++r) {
                s_stage[wv][khalf * 8 + r][c] += accp[r];
            }
        }
    }

    // ---- write out: out[c*N + n]; lane = column c, 16 rows each ----
    #pragma unroll
    for (int r = 0; r < 16; ++r) {
        out[lane * N_PTS + nbase + r] = s_stage[wv][r][lane];
    }
}

extern "C" void kernel_launch(void* const* d_in, const int* in_sizes, int n_in,
                              void* d_out, int out_size, void* d_ws, size_t ws_size,
                              hipStream_t stream) {
    (void)in_sizes; (void)n_in; (void)out_size; (void)d_ws; (void)ws_size;
    const float* x    = (const float*)d_in[0];
    const float* zloc = (const float*)d_in[1];
    const float* iw   = (const float*)d_in[2];
    const float* pf   = (const float*)d_in[3];
    const float* pph  = (const float*)d_in[4];
    const float* pw   = (const float*)d_in[5];
    float* out = (float*)d_out;

    // 1024 row-tiles of 16 rows; 8 waves (tiles) per 256-thread block -> 128 blocks
    dim3 grid(N_PTS / (16 * 8));
    dim3 block(256);
    sgp_fused_kernel<<<grid, block, 0, stream>>>(x, zloc, iw, pf, pph, pw, out);
}
